// DeformableTransformer_4209067950222
// MI455X (gfx1250) — compile-verified, matrix operand order checked by reading, not run on hardware
//
#include <hip/hip_runtime.h>
#include <hip/hip_bf16.h>
#include <math.h>
#include <stdint.h>

// ---------------------------------------------------------------------------
// Deformable DETR decoder (6 layers) for MI455X / gfx1250 (wave32, WMMA).
//  * all GEMM operands converted to f16 once in global memory (halves HBM
//    traffic; 23.3 TB/s chip, GEMM data must stream at low precision)
//  * GEMM: 128x64 block tile, 32x32 per wave (4 x v_wmma_f32_16x16x32_f16
//    per BK=32 step), double-buffered LDS tiles filled with
//    GLOBAL_LOAD_ASYNC_TO_LDS_B128 (ASYNCcnt-tracked CDNA5 async copies)
//  * self-attention: flash-style online softmax, WMMA for S=Q*K^T and P*V,
//    16-lane softmax reductions done with DPP (row = 16 lanes) instead of
//    ds_bpermute round-trips
//  * deformable sampling: one wave per (b,q,head), lane = channel, f16 gathers
// ---------------------------------------------------------------------------

typedef __attribute__((ext_vector_type(16))) _Float16 v16h;
typedef __attribute__((ext_vector_type(8)))  _Float16 v8h;
typedef __attribute__((ext_vector_type(4)))  _Float16 v4h;
typedef __attribute__((ext_vector_type(8)))  float    v8f;
// async builtin parameter types (per hipcc diagnostic): vector-of-int pointers
typedef int vsi4 __attribute__((vector_size(16)));
typedef int vsi2 __attribute__((vector_size(8)));
typedef __attribute__((address_space(1))) vsi4 gvsi4;
typedef __attribute__((address_space(3))) vsi4 lvsi4;
typedef __attribute__((address_space(1))) vsi2 gvsi2;
typedef __attribute__((address_space(3))) vsi2 lvsi2;

#define C_DIM   256
#define NHEAD   8
#define DH      32
#define NLVL    4
#define NPT     4
#define DFF     1024
#define NLAYERS 6
#define BATCH   4
#define LQ      900
#define LEN_IN  13294
#define M_Q     (BATCH * LQ)       // 3600
#define M_V     (BATCH * LEN_IN)   // 53176
#define LN_EPS  1e-5f
#define LDT     40                 // LDS row stride in halves (80B, 16B aligned)

// ---- CDNA5 async global->LDS copy (guarded; falls back to VGPR copies) ----
#if __has_builtin(__builtin_amdgcn_global_load_async_to_lds_b128) && \
    __has_builtin(__builtin_amdgcn_global_load_async_to_lds_b64) &&  \
    __has_builtin(__builtin_amdgcn_s_wait_asynccnt)
#define USE_ASYNC_LDS 1
#else
#define USE_ASYNC_LDS 0
#endif

__device__ inline void async_copy_b128(const _Float16* g, _Float16* l) {
#if USE_ASYNC_LDS
  __builtin_amdgcn_global_load_async_to_lds_b128(
      (gvsi4*)(uintptr_t)g, (lvsi4*)(uint32_t)(uintptr_t)l, 0, 0);
#else
  *(v8h*)l = *(const v8h*)g;
#endif
}

__device__ inline void async_copy_b64(const _Float16* g, _Float16* l) {
#if USE_ASYNC_LDS
  __builtin_amdgcn_global_load_async_to_lds_b64(
      (gvsi2*)(uintptr_t)g, (lvsi2*)(uint32_t)(uintptr_t)l, 0, 0);
#else
  *(v4h*)l = *(const v4h*)g;
#endif
}

// wait until at most `keep3` (true: 3, false: 0) async copies outstanding
__device__ inline void wait_async(bool keep3) {
#if USE_ASYNC_LDS
  if (keep3) __builtin_amdgcn_s_wait_asynccnt(3);
  else       __builtin_amdgcn_s_wait_asynccnt(0);
#else
  (void)keep3;
#endif
}

// ---- DPP 16-lane (one DPP row) butterfly reductions: pure VALU, no LDS ----
#if __has_builtin(__builtin_amdgcn_update_dpp)
#define USE_DPP 1
template <int CTRL>
__device__ inline float dpp_mov(float v) {
  return __builtin_bit_cast(
      float, __builtin_amdgcn_update_dpp(0, __builtin_bit_cast(int, v), CTRL,
                                         0xf, 0xf, true));
}
#else
#define USE_DPP 0
#endif

__device__ inline float row16_max(float v) {
#if USE_DPP
  v = fmaxf(v, dpp_mov<0xB1>(v));   // quad_perm(1,0,3,2)  : xor 1
  v = fmaxf(v, dpp_mov<0x4E>(v));   // quad_perm(2,3,0,1)  : xor 2
  v = fmaxf(v, dpp_mov<0x141>(v));  // row_half_mirror     (quads uniform)
  v = fmaxf(v, dpp_mov<0x140>(v));  // row_mirror          (8-groups uniform)
#else
  for (int sh = 1; sh < 16; sh <<= 1) v = fmaxf(v, __shfl_xor(v, sh, 32));
#endif
  return v;
}

__device__ inline float row16_sum(float v) {
#if USE_DPP
  v += dpp_mov<0xB1>(v);
  v += dpp_mov<0x4E>(v);
  v += dpp_mov<0x141>(v);
  v += dpp_mov<0x140>(v);
#else
  for (int sh = 1; sh < 16; sh <<= 1) v += __shfl_xor(v, sh, 32);
#endif
  return v;
}

// Load a 16x32 f16 fragment in WMMA A/B per-lane layout from an LDS tile
// stored row-major (rows x 32 halves, stride ld halves, 16B aligned).
// Lane L: row = row0 + (L & 15); elements i<8 -> k = kb+i, i>=8 -> k = kb+16+i-8,
// kb = 8 for lanes 16..31 (CDNA5 ISA 16-bit 16x32 A-matrix layout; B mirrors
// it with rows = N since S/D layouts stripe rows across lanes).
__device__ inline v16h ldsFrag(const _Float16* base, int row0, int lane, int ld) {
  const _Float16* p = base + (size_t)(row0 + (lane & 15)) * ld + ((lane >> 4) << 3);
  v8h lo = *(const v8h*)p;
  v8h hi = *(const v8h*)(p + 16);
  v16h f;
#pragma unroll
  for (int i = 0; i < 8; ++i) { f[i] = lo[i]; f[i + 8] = hi[i]; }
  return f;
}

// ---------------------------------------------------------------------------
// Y(M,N) = Xh(M,K) @ Wh(N,K)^T + bias   [optional ReLU; f16 or f32 output]
// 256 threads = 8 waves. Block tile 128x64, BK=32, double-buffered LDS.
// Wave w: rows (w>>1)*32, cols (w&1)*32 -> 4 WMMA accumulators (32x32).
// Tail M rows are clamped at load (their outputs are store-guarded), so each
// thread issues exactly 3 async copies per stage -> s_wait_asynccnt 3 pipelining.
// ---------------------------------------------------------------------------
__device__ inline void stage_tiles(const _Float16* __restrict__ X,
                                   const _Float16* __restrict__ W,
                                   _Float16* sA, _Float16* sB,
                                   int mTile, int nTile, int M, int K, int k0,
                                   int tid) {
#pragma unroll
  for (int u = 0; u < 2; ++u) {                  // A: 128 rows x 4 b128 groups
    const int idx = tid + u * 256;
    const int row = idx >> 2, grp = (idx & 3) << 3;
    int gr = mTile + row;
    if (gr >= M) gr = M - 1;                     // clamp (outputs guarded)
    async_copy_b128(X + (size_t)gr * K + k0 + grp, sA + row * LDT + grp);
  }
  {                                              // B: 64 rows x 4 b128 groups
    const int row = tid >> 2, grp = (tid & 3) << 3;
    async_copy_b128(W + (size_t)(nTile + row) * K + k0 + grp, sB + row * LDT + grp);
  }
}

template <int OUT_F16>
__global__ __launch_bounds__(256)
void gemm_wmma_kernel(const _Float16* __restrict__ X, const _Float16* __restrict__ W,
                      const float* __restrict__ bias, void* __restrict__ Yv,
                      int M, int N, int K, int relu) {
  __shared__ __align__(16) _Float16 sA[2][128 * LDT];
  __shared__ __align__(16) _Float16 sB[2][64 * LDT];
  const int tid  = threadIdx.x;
  const int lane = tid & 31, wid = tid >> 5;
  const int mTile = blockIdx.y * 128, nTile = blockIdx.x * 64;
  const int mRow = (wid >> 1) * 32, nCol = (wid & 1) * 32;
  v8f c00 = {}; v8f c01 = {}; v8f c10 = {}; v8f c11 = {};

  const int nK = K >> 5;
  stage_tiles(X, W, sA[0], sB[0], mTile, nTile, M, K, 0, tid);
  for (int kt = 0; kt < nK; ++kt) {
    const int cur = kt & 1;
    const bool more = (kt + 1) < nK;
    if (more)
      stage_tiles(X, W, sA[cur ^ 1], sB[cur ^ 1], mTile, nTile, M, K,
                  (kt + 1) << 5, tid);
    wait_async(more);          // this wave's stage(cur) copies complete
    __syncthreads();           // ... and everyone else's

    const v16h a0 = ldsFrag(sA[cur], mRow,      lane, LDT);
    const v16h a1 = ldsFrag(sA[cur], mRow + 16, lane, LDT);
    const v16h b0 = ldsFrag(sB[cur], nCol,      lane, LDT);
    const v16h b1 = ldsFrag(sB[cur], nCol + 16, lane, LDT);
    c00 = __builtin_amdgcn_wmma_f32_16x16x32_f16(false, a0, false, b0, (short)0, c00, false, false);
    c01 = __builtin_amdgcn_wmma_f32_16x16x32_f16(false, a0, false, b1, (short)0, c01, false, false);
    c10 = __builtin_amdgcn_wmma_f32_16x16x32_f16(false, a1, false, b0, (short)0, c10, false, false);
    c11 = __builtin_amdgcn_wmma_f32_16x16x32_f16(false, a1, false, b1, (short)0, c11, false, false);
    __syncthreads();           // done reading buf(cur) before it is restaged
  }

  // C/D layout: lane n = lane&15; VGPR r -> row r (+8 for lanes 16..31)
  const int n0 = nTile + nCol + (lane & 15);
  const int mB = mTile + mRow + ((lane >> 4) << 3);
  const float bv0 = bias[n0], bv1 = bias[n0 + 16];
#pragma unroll
  for (int r = 0; r < 8; ++r) {
    float y00 = c00[r] + bv0, y01 = c01[r] + bv1;
    float y10 = c10[r] + bv0, y11 = c11[r] + bv1;
    if (relu) {
      y00 = fmaxf(y00, 0.f); y01 = fmaxf(y01, 0.f);
      y10 = fmaxf(y10, 0.f); y11 = fmaxf(y11, 0.f);
    }
    const int m0 = mB + r, m1 = mB + 16 + r;
    if (m0 < M) {
      if (OUT_F16) {
        ((_Float16*)Yv)[(size_t)m0 * N + n0]      = (_Float16)y00;
        ((_Float16*)Yv)[(size_t)m0 * N + n0 + 16] = (_Float16)y01;
      } else {
        ((float*)Yv)[(size_t)m0 * N + n0]      = y00;
        ((float*)Yv)[(size_t)m0 * N + n0 + 16] = y01;
      }
    }
    if (m1 < M) {
      if (OUT_F16) {
        ((_Float16*)Yv)[(size_t)m1 * N + n0]      = (_Float16)y10;
        ((_Float16*)Yv)[(size_t)m1 * N + n0 + 16] = (_Float16)y11;
      } else {
        ((float*)Yv)[(size_t)m1 * N + n0]      = y10;
        ((float*)Yv)[(size_t)m1 * N + n0 + 16] = y11;
      }
    }
  }
}

// ---------------------------------------------------------------------------
// Flash attention, f16 in / f16 out. grid = (ceil(LQ/128), B*NHEAD).
// Each wave owns a 16-row Q tile; 32-key K/V tiles staged in LDS.
// K tile filled with async b64 copies (clamped key rows; masked columns are
// overwritten after WMMA, so clamp is safe). V is transposed via VGPRs.
// Softmax row reductions use DPP within each 16-lane half-wave.
// ---------------------------------------------------------------------------
__global__ __launch_bounds__(256)
void flash_attn_kernel(const _Float16* __restrict__ Qh, const _Float16* __restrict__ Kh,
                       const _Float16* __restrict__ Vh, _Float16* __restrict__ O) {
  __shared__ __align__(16) _Float16 sK [32 * LDT];      // [key][dh]
  __shared__ __align__(16) _Float16 sVt[32 * LDT];      // [dh][key]  (V^T)
  __shared__ __align__(16) _Float16 sP [8][16 * LDT];   // per-wave P staging
  const int tid = threadIdx.x, lane = tid & 31, wid = tid >> 5;
  const int b = blockIdx.y >> 3, h = blockIdx.y & 7;
  const int qBase = blockIdx.x * 128 + wid * 16;
  const float scl = 0.17677669529663687f;               // 1/sqrt(DH)

  v16h aq;                                              // Q fragment, pre-scaled
  {
    const int m  = qBase + (lane & 15);
    const int kb = (lane >> 4) << 3;
    const _Float16* qp =
        Qh + ((size_t)b * LQ + (m < LQ ? m : (LQ - 1))) * C_DIM + h * DH;
    const float gate = (m < LQ) ? scl : 0.f;
#pragma unroll
    for (int i = 0; i < 8; ++i) {
      aq[i]     = (_Float16)((float)qp[kb + i] * gate);
      aq[i + 8] = (_Float16)((float)qp[kb + 16 + i] * gate);
    }
  }

  v8f o0 = {}; v8f o1 = {};
  float rmax[8], rsum[8];
#pragma unroll
  for (int r = 0; r < 8; ++r) { rmax[r] = -1e30f; rsum[r] = 0.f; }

  for (int kt = 0; kt < LQ; kt += 32) {
    __syncthreads();                                    // sK/sVt free for reuse
    {
      const int key = tid >> 3;                         // 0..31
      const int d0  = (tid & 7) << 2;                   // 0,4,..,28
      int gk = kt + key;
      if (gk >= LQ) gk = LQ - 1;                        // clamp (cols masked)
      const size_t roff = ((size_t)b * LQ + gk) * C_DIM + h * DH + d0;
      async_copy_b64(Kh + roff, sK + key * LDT + d0);   // CDNA5 async copy
      const v4h vv = *(const v4h*)(Vh + roff);          // transpose via VGPRs
      sVt[(d0 + 0) * LDT + key] = vv[0];
      sVt[(d0 + 1) * LDT + key] = vv[1];
      sVt[(d0 + 2) * LDT + key] = vv[2];
      sVt[(d0 + 3) * LDT + key] = vv[3];
    }
    wait_async(false);
    __syncthreads();

    const v16h bk0 = ldsFrag(sK, 0,  lane, LDT);
    const v16h bk1 = ldsFrag(sK, 16, lane, LDT);
    v8f s0 = {}; v8f s1 = {};
    s0 = __builtin_amdgcn_wmma_f32_16x16x32_f16(false, aq, false, bk0, (short)0, s0, false, false);
    s1 = __builtin_amdgcn_wmma_f32_16x16x32_f16(false, aq, false, bk1, (short)0, s1, false, false);

    const int j0 = kt + (lane & 15), j1 = kt + 16 + (lane & 15);
#pragma unroll
    for (int r = 0; r < 8; ++r) {
      float p0 = (j0 < LQ) ? s0[r] : -1e30f;
      float p1 = (j1 < LQ) ? s1[r] : -1e30f;
      const float tm   = row16_max(fmaxf(p0, p1));   // DPP row reduce
      const float nm   = fmaxf(rmax[r], tm);
      const float corr = __expf(rmax[r] - nm);
      const float e0 = __expf(p0 - nm);
      const float e1 = __expf(p1 - nm);
      const float ts = row16_sum(e0 + e1);           // DPP row reduce
      rsum[r] = rsum[r] * corr + ts;
      o0[r] *= corr; o1[r] *= corr;
      rmax[r] = nm;
      const int m = r + ((lane >> 4) << 3);  // stage P: C layout -> row major
      sP[wid][m * LDT + (lane & 15)]      = (_Float16)e0;
      sP[wid][m * LDT + 16 + (lane & 15)] = (_Float16)e1;
    }
    // same-wave LDS ops are in-order; no barrier needed for private sP slab
    const v16h ap  = ldsFrag(sP[wid], 0, lane, LDT);
    const v16h bv0 = ldsFrag(sVt, 0,  lane, LDT);
    const v16h bv1 = ldsFrag(sVt, 16, lane, LDT);
    o0 = __builtin_amdgcn_wmma_f32_16x16x32_f16(false, ap, false, bv0, (short)0, o0, false, false);
    o1 = __builtin_amdgcn_wmma_f32_16x16x32_f16(false, ap, false, bv1, (short)0, o1, false, false);
  }

  const int n = lane & 15;
  const int mBase = qBase + ((lane >> 4) << 3);
#pragma unroll
  for (int r = 0; r < 8; ++r) {
    const int m = mBase + r;
    if (m < LQ) {
      const float inv = 1.f / rsum[r];
      _Float16* op = O + ((size_t)b * LQ + m) * C_DIM + h * DH;
      op[n]      = (_Float16)(o0[r] * inv);
      op[n + 16] = (_Float16)(o1[r] * inv);
    }
  }
}

// ---------------------------------------------------------------------------
// Multi-scale deformable sampling: one wave per (b,q,head); lane = channel.
// f16 value gathers (64B per corner per wave), f32 math, f16 output.
// ---------------------------------------------------------------------------
__global__ __launch_bounds__(256)
void deform_attn_kernel(const float* __restrict__ offs, const float* __restrict__ awr,
                        const float* __restrict__ refp, const float* __restrict__ vratio,
                        const _Float16* __restrict__ value, _Float16* __restrict__ out) {
  const int gw   = (blockIdx.x * blockDim.x + threadIdx.x) >> 5;
  const int lane = threadIdx.x & 31;
  if (gw >= BATCH * LQ * NHEAD) return;
  const int h = gw % NHEAD;
  const int q = (gw / NHEAD) % LQ;
  const int b = gw / (NHEAD * LQ);

  const int lvlH[NLVL] = {100, 50, 25, 13};
  const int lvlW[NLVL] = {100, 50, 25, 13};
  const int lvlS[NLVL] = {0, 10000, 12500, 13125};

  const float* ap = awr + ((size_t)b * LQ + q) * (NHEAD * NLVL * NPT) + h * NLVL * NPT;
  float aw[16]; float amax = -1e30f;
#pragma unroll
  for (int j = 0; j < 16; ++j) { aw[j] = ap[j]; amax = fmaxf(amax, aw[j]); }
  float asum = 0.f;
#pragma unroll
  for (int j = 0; j < 16; ++j) { aw[j] = __expf(aw[j] - amax); asum += aw[j]; }
  const float ainv = 1.f / asum;

  const float rx = refp[((size_t)b * LQ + q) * 2 + 0];
  const float ry = refp[((size_t)b * LQ + q) * 2 + 1];
  const float* op = offs + ((size_t)b * LQ + q) * (NHEAD * NLVL * NPT * 2) + h * NLVL * NPT * 2;

  float acc = 0.f;
#pragma unroll
  for (int l = 0; l < NLVL; ++l) {
    const int hl = lvlH[l], wl = lvlW[l];
    const float cx = rx * vratio[(b * NLVL + l) * 2 + 0];
    const float cy = ry * vratio[(b * NLVL + l) * 2 + 1];
    const _Float16* vbase =
        value + ((size_t)b * LEN_IN + lvlS[l]) * C_DIM + h * DH + lane;
#pragma unroll
    for (int p = 0; p < NPT; ++p) {
      const float ox = op[(l * NPT + p) * 2 + 0];
      const float oy = op[(l * NPT + p) * 2 + 1];
      const float x = (cx + ox / (float)wl) * (float)wl - 0.5f;
      const float y = (cy + oy / (float)hl) * (float)hl - 0.5f;
      const float x0f = floorf(x), y0f = floorf(y);
      const float lx = x - x0f, ly = y - y0f;
      const int x0 = (int)x0f, y0 = (int)y0f;
      float s = 0.f;
#pragma unroll
      for (int cyi = 0; cyi < 2; ++cyi) {
#pragma unroll
        for (int cxi = 0; cxi < 2; ++cxi) {
          const int yi = y0 + cyi, xi = x0 + cxi;
          const float cw = (cxi ? lx : 1.f - lx) * (cyi ? ly : 1.f - ly);
          if (yi >= 0 && yi < hl && xi >= 0 && xi < wl)
            s += cw * (float)vbase[(size_t)(yi * wl + xi) * C_DIM];
        }
      }
      acc += (aw[l * NPT + p] * ainv) * s;
    }
  }
  out[((size_t)b * LQ + q) * C_DIM + h * DH + lane] = (_Float16)acc;
}

// ---------------------------------------------------------------------------
// x = LayerNorm(x + t2) * g + b ; also writes the f16 mirror xh of x.
// ---------------------------------------------------------------------------
__global__ __launch_bounds__(256)
void add_ln_kernel(float* __restrict__ x, _Float16* __restrict__ xh,
                   const float* __restrict__ t2,
                   const float* __restrict__ g, const float* __restrict__ bta) {
  __shared__ float ssum[8], ssq[8];
  const int row = blockIdx.x, tid = threadIdx.x;
  const size_t base = (size_t)row * C_DIM;
  const float v = x[base + tid] + t2[base + tid];
  float s = v, s2 = v * v;
  s  = row16_sum(s);   s  += __shfl_xor(s,  16, 32);
  s2 = row16_sum(s2);  s2 += __shfl_xor(s2, 16, 32);
  if ((tid & 31) == 0) { ssum[tid >> 5] = s; ssq[tid >> 5] = s2; }
  __syncthreads();
  float tot = 0.f, tot2 = 0.f;
#pragma unroll
  for (int i = 0; i < 8; ++i) { tot += ssum[i]; tot2 += ssq[i]; }
  const float mean = tot * (1.f / C_DIM);
  const float var  = tot2 * (1.f / C_DIM) - mean * mean;
  const float rstd = rsqrtf(var + LN_EPS);
  const float y = (v - mean) * rstd * g[tid] + bta[tid];
  x[base + tid]  = y;
  xh[base + tid] = (_Float16)y;
}

__global__ __launch_bounds__(256)
void add_f16_kernel(const float* __restrict__ a, const float* __restrict__ b,
                    _Float16* __restrict__ o, int n) {
  const int i = blockIdx.x * blockDim.x + threadIdx.x;
  if (i < n) o[i] = (_Float16)(a[i] + b[i]);
}

__global__ __launch_bounds__(256)
void cvt_f16_kernel(const float* __restrict__ in, _Float16* __restrict__ out, int n) {
  const int i = blockIdx.x * blockDim.x + threadIdx.x;
  if (i < n) out[i] = (_Float16)in[i];
}

// ---------------------------------------------------------------------------
// Host-side orchestration
// ---------------------------------------------------------------------------
extern "C" void kernel_launch(void* const* d_in, const int* in_sizes, int n_in,
                              void* d_out, int out_size, void* d_ws, size_t ws_size,
                              hipStream_t stream) {
  (void)in_sizes; (void)n_in; (void)out_size; (void)ws_size;
  const float* tgt    = (const float*)d_in[0];
  const float* qpos   = (const float*)d_in[1];
  const float* refp   = (const float*)d_in[2];
  const float* src    = (const float*)d_in[3];
  const float* vratio = (const float*)d_in[4];

  uint8_t* base = (uint8_t*)d_ws;
  size_t off = 0;
  auto alloc = [&](size_t bytes) -> void* {
    void* p = base + off;
    off = (off + bytes + 255) & ~(size_t)255;
    return p;
  };
  float*    x    = (float*)   alloc((size_t)M_Q * C_DIM * 4);
  float*    t2B  = (float*)   alloc((size_t)M_Q * C_DIM * 4);
  float*    offsB= (float*)   alloc((size_t)M_Q * C_DIM * 4);
  float*    awrB = (float*)   alloc((size_t)M_Q * 128 * 4);
  _Float16* xh   = (_Float16*)alloc((size_t)M_Q * C_DIM * 2);
  _Float16* qbh  = (_Float16*)alloc((size_t)M_Q * C_DIM * 2);
  _Float16* qh   = (_Float16*)alloc((size_t)M_Q * C_DIM * 2);
  _Float16* kh   = (_Float16*)alloc((size_t)M_Q * C_DIM * 2);
  _Float16* vh   = (_Float16*)alloc((size_t)M_Q * C_DIM * 2);
  _Float16* aoh  = (_Float16*)alloc((size_t)M_Q * C_DIM * 2);
  _Float16* hidh = (_Float16*)alloc((size_t)M_Q * DFF * 2);
  _Float16* srch = (_Float16*)alloc((size_t)M_V * C_DIM * 2);
  _Float16* valh = (_Float16*)alloc((size_t)M_V * C_DIM * 2);
  // per-layer f16 weight staging
  _Float16* w_attn_in = (_Float16*)alloc((size_t)768 * 256 * 2);
  _Float16* w_attn_out= (_Float16*)alloc((size_t)256 * 256 * 2);
  _Float16* w_off     = (_Float16*)alloc((size_t)256 * 256 * 2);
  _Float16* w_aw      = (_Float16*)alloc((size_t)128 * 256 * 2);
  _Float16* w_val     = (_Float16*)alloc((size_t)256 * 256 * 2);
  _Float16* w_out     = (_Float16*)alloc((size_t)256 * 256 * 2);
  _Float16* w_ffn1    = (_Float16*)alloc((size_t)1024 * 256 * 2);
  _Float16* w_ffn2    = (_Float16*)alloc((size_t)256 * 1024 * 2);

  const dim3 blk(256);
  auto cvt = [&](const float* in, _Float16* out, size_t n) {
    cvt_f16_kernel<<<dim3((n + 255) / 256), blk, 0, stream>>>(in, out, (int)n);
  };
  auto gemm32 = [&](const _Float16* X, const _Float16* W, const float* bias,
                    float* Y, int M, int N, int K, int relu) {
    gemm_wmma_kernel<0><<<dim3(N / 64, (M + 127) / 128), blk, 0, stream>>>(
        X, W, bias, (void*)Y, M, N, K, relu);
  };
  auto gemm16 = [&](const _Float16* X, const _Float16* W, const float* bias,
                    _Float16* Y, int M, int N, int K, int relu) {
    gemm_wmma_kernel<1><<<dim3(N / 64, (M + 127) / 128), blk, 0, stream>>>(
        X, W, bias, (void*)Y, M, N, K, relu);
  };

  (void)hipMemcpyAsync(x, tgt, (size_t)M_Q * C_DIM * sizeof(float),
                       hipMemcpyDeviceToDevice, stream);
  cvt(tgt, xh, (size_t)M_Q * C_DIM);
  cvt(src, srch, (size_t)M_V * C_DIM);

  for (int l = 0; l < NLAYERS; ++l) {
    auto prm = [&](int i) { return (const float*)d_in[7 + l * 22 + i]; };
    const float* attn_in_b  = prm(1);
    const float* attn_out_b = prm(3);
    const float* off_b  = prm(5);
    const float* aw_b   = prm(7);
    const float* val_b  = prm(9);
    const float* out_b  = prm(11);
    const float* ffn1_b = prm(13);
    const float* ffn2_b = prm(15);
    const float* ln1_g  = prm(16); const float* ln1_b  = prm(17);
    const float* ln2_g  = prm(18); const float* ln2_b  = prm(19);
    const float* ln3_g  = prm(20); const float* ln3_b  = prm(21);

    cvt(prm(0),  w_attn_in,  (size_t)768 * 256);
    cvt(prm(2),  w_attn_out, (size_t)256 * 256);
    cvt(prm(4),  w_off,      (size_t)256 * 256);
    cvt(prm(6),  w_aw,       (size_t)128 * 256);
    cvt(prm(8),  w_val,      (size_t)256 * 256);
    cvt(prm(10), w_out,      (size_t)256 * 256);
    cvt(prm(12), w_ffn1,     (size_t)1024 * 256);
    cvt(prm(14), w_ffn2,     (size_t)256 * 1024);

    // ---- self attention: q = k = x + qpos, v = x ----
    add_f16_kernel<<<dim3(M_Q * C_DIM / 256), blk, 0, stream>>>(
        x, qpos, qbh, M_Q * C_DIM);
    gemm16(qbh, w_attn_in,              attn_in_b,       qh, M_Q, C_DIM, C_DIM, 0);
    gemm16(qbh, w_attn_in + 256 * 256,  attn_in_b + 256, kh, M_Q, C_DIM, C_DIM, 0);
    gemm16(xh,  w_attn_in + 512 * 256,  attn_in_b + 512, vh, M_Q, C_DIM, C_DIM, 0);
    flash_attn_kernel<<<dim3((LQ + 127) / 128, BATCH * NHEAD), blk, 0, stream>>>(
        qh, kh, vh, aoh);
    gemm32(aoh, w_attn_out, attn_out_b, t2B, M_Q, C_DIM, C_DIM, 0);
    add_ln_kernel<<<dim3(M_Q), blk, 0, stream>>>(x, xh, t2B, ln2_g, ln2_b);

    // ---- deformable attention ----
    add_f16_kernel<<<dim3(M_Q * C_DIM / 256), blk, 0, stream>>>(
        x, qpos, qbh, M_Q * C_DIM);
    gemm16(srch, w_val, val_b, valh, M_V, C_DIM, C_DIM, 0);
    gemm32(qbh, w_off, off_b, offsB, M_Q, 256, C_DIM, 0);
    gemm32(qbh, w_aw,  aw_b,  awrB,  M_Q, 128, C_DIM, 0);
    deform_attn_kernel<<<dim3(BATCH * LQ * NHEAD / 8), blk, 0, stream>>>(
        offsB, awrB, refp, vratio, valh, aoh);
    gemm32(aoh, w_out, out_b, t2B, M_Q, C_DIM, C_DIM, 0);
    add_ln_kernel<<<dim3(M_Q), blk, 0, stream>>>(x, xh, t2B, ln1_g, ln1_b);

    // ---- FFN ----
    gemm16(xh,   w_ffn1, ffn1_b, hidh, M_Q, DFF,   C_DIM, 1);
    gemm32(hidh, w_ffn2, ffn2_b, t2B,  M_Q, C_DIM, DFF,   0);
    add_ln_kernel<<<dim3(M_Q), blk, 0, stream>>>(x, xh, t2B, ln3_g, ln3_b);
  }

  // tuple output: (out, reference_points)
  (void)hipMemcpyAsync(d_out, x, (size_t)M_Q * C_DIM * sizeof(float),
                       hipMemcpyDeviceToDevice, stream);
  (void)hipMemcpyAsync((float*)d_out + (size_t)M_Q * C_DIM, refp,
                       (size_t)BATCH * LQ * 2 * sizeof(float),
                       hipMemcpyDeviceToDevice, stream);
}
